// Attention_46222438040305
// MI455X (gfx1250) — matrix operand order repaired and checked by reading, not compiled, not run on hardware
//
#include <hip/hip_runtime.h>

#define DM   1024
#define NH   16
#define DH   64
#define BATCH 4
#define SEQ  2048

typedef __attribute__((ext_vector_type(16))) __bf16 v16bf;
typedef __attribute__((ext_vector_type(8)))  __bf16 v8bf;
typedef __attribute__((ext_vector_type(8)))  float  v8f;

static __device__ __forceinline__ v16bf cat16(v8bf a, v8bf b) {
  return __builtin_shufflevector(a, b, 0,1,2,3,4,5,6,7,8,9,10,11,12,13,14,15);
}
static __device__ __forceinline__ v8f wmma_bf16(v16bf a, v16bf b, v8f c) {
  return __builtin_amdgcn_wmma_f32_16x16x32_bf16(false, a, false, b, (short)0, c, false, false);
}

// ---- async global->LDS copy (16B per lane), with portable fallback ----------
#if __has_builtin(__builtin_amdgcn_global_load_async_to_lds_b128)
#define HAS_ASYNC_LDS 1
#else
#define HAS_ASYNC_LDS 0
#endif

typedef int v4i_ __attribute__((vector_size(16)));
typedef __attribute__((address_space(1))) v4i_* as1_v4i;
typedef __attribute__((address_space(3))) v4i_* as3_v4i;

static __device__ __forceinline__ void cp16_g2l(const __bf16* g, __bf16* l) {
#if HAS_ASYNC_LDS
  __builtin_amdgcn_global_load_async_to_lds_b128((as1_v4i)g, (as3_v4i)l, 0, 0);
#else
  *(uint4*)l = *(const uint4*)g;
#endif
}
static __device__ __forceinline__ void wait_async() {
#if HAS_ASYNC_LDS
#if __has_builtin(__builtin_amdgcn_s_wait_asynccnt)
  __builtin_amdgcn_s_wait_asynccnt(0);
#else
  asm volatile("s_wait_asynccnt 0" ::: "memory");
#endif
#endif
}

// ---------------------------------------------------------------- prep
// WqT/WkT/WvT[h][d][m] = W[h][m][d] (bf16);  WoT[mo][hd] = W_O[h][d][mo] (bf16)
__global__ __launch_bounds__(256) void prep_weights(
    const float* __restrict__ Wq, const float* __restrict__ Wk,
    const float* __restrict__ Wv, const float* __restrict__ Wo,
    __bf16* __restrict__ WqT, __bf16* __restrict__ WkT,
    __bf16* __restrict__ WvT, __bf16* __restrict__ WoT) {
  const int NW = NH * DM * DH;                 // 1,048,576 per projection matrix
  int i = blockIdx.x * 256 + threadIdx.x;
  if (i < 3 * NW) {
    int mat = i / NW, r = i % NW;
    int h = r / (DH * DM), r2 = r % (DH * DM);
    int d = r2 / DM, m = r2 % DM;
    const float* W = (mat == 0) ? Wq : (mat == 1) ? Wk : Wv;
    __bf16* Wt     = (mat == 0) ? WqT : (mat == 1) ? WkT : WvT;
    Wt[((size_t)h * DH + d) * DM + m] = (__bf16)W[((size_t)h * DM + m) * DH + d];
  } else {
    int r = i - 3 * NW;
    if (r < DM * DM) {
      int mo = r / DM, hd = r % DM;            // hd = h*64+d
      WoT[(size_t)mo * DM + hd] = (__bf16)Wo[(size_t)hd * DM + mo];
    }
  }
}

// ---------------------------------------------------------------- QKV projection
// grid (B*S/16, 6), block 256 (8 waves). Wave -> one (matrix, head) combo.
// Ping-pong double buffering (unroll-by-2, no register copies).
__global__ __launch_bounds__(256) void qkv_proj(
    const float* __restrict__ x,
    const __bf16* __restrict__ WqT, const __bf16* __restrict__ WkT,
    const __bf16* __restrict__ WvT,
    const float* __restrict__ bQ, const float* __restrict__ bK,
    const float* __restrict__ bV,
    __bf16* __restrict__ Qb, __bf16* __restrict__ Kb, __bf16* __restrict__ Vt) {
  __shared__ alignas(16) __bf16 xs[16 * DM];     // 32 KB x-tile in bf16
  const int tid = threadIdx.x;
  const int bs = blockIdx.x;
  const int b = bs / (SEQ / 16);
  const int s_base = (bs % (SEQ / 16)) * 16;

  const float* xt = x + ((size_t)b * SEQ + s_base) * DM;
  for (int i = tid * 4; i < 16 * DM; i += 256 * 4) {
    float4 v = *(const float4*)(xt + i);
    xs[i + 0] = (__bf16)v.x; xs[i + 1] = (__bf16)v.y;
    xs[i + 2] = (__bf16)v.z; xs[i + 3] = (__bf16)v.w;
  }
  __syncthreads();

  const int wave = tid >> 5, lane = tid & 31;
  const int combo = blockIdx.y * 8 + wave;       // 0..47
  const int mat = combo / NH, h = combo % NH;
  const __bf16* Wt  = ((mat == 0) ? WqT : (mat == 1) ? WkT : WvT) + (size_t)h * DH * DM;
  const float* bias = ((mat == 0) ? bQ  : (mat == 1) ? bK  : bV ) + h * DH;

  const int half = lane >> 4, row = lane & 15;
  const int selA = half * 8, selB = half * 8;
  const __bf16* xrow = xs + row * DM;

  auto loadA = [&](int ks) -> v16bf {
    const int mb = ks * 32;
    return cat16(*(const v8bf*)(xrow + mb + selA),
                 *(const v8bf*)(xrow + mb + 16 + selA));
  };
  auto loadB = [&](int ks, int t) -> v16bf {
    const __bf16* wp = Wt + (size_t)(t * 16 + row) * DM + ks * 32 + selB;
    return cat16(*(const v8bf*)wp, *(const v8bf*)(wp + 16));
  };

  constexpr int NK = DM / 32;                    // 32 k-steps (even)
  v8f acc[4] = {};
  v16bf a0 = loadA(0), a1;
  v16bf b0[4], b1[4];
#pragma unroll
  for (int t = 0; t < 4; ++t) b0[t] = loadB(0, t);

  for (int ks = 0; ks < NK - 2; ks += 2) {
    a1 = loadA(ks + 1);
#pragma unroll
    for (int t = 0; t < 4; ++t) b1[t] = loadB(ks + 1, t);
#pragma unroll
    for (int t = 0; t < 4; ++t) acc[t] = wmma_bf16(a0, b0[t], acc[t]);
    a0 = loadA(ks + 2);
#pragma unroll
    for (int t = 0; t < 4; ++t) b0[t] = loadB(ks + 2, t);
#pragma unroll
    for (int t = 0; t < 4; ++t) acc[t] = wmma_bf16(a1, b1[t], acc[t]);
  }
  // tail: ks = NK-2, NK-1
  a1 = loadA(NK - 1);
#pragma unroll
  for (int t = 0; t < 4; ++t) b1[t] = loadB(NK - 1, t);
#pragma unroll
  for (int t = 0; t < 4; ++t) acc[t] = wmma_bf16(a0, b0[t], acc[t]);
#pragma unroll
  for (int t = 0; t < 4; ++t) acc[t] = wmma_bf16(a1, b1[t], acc[t]);

#pragma unroll
  for (int t = 0; t < 4; ++t) {
    const int d = t * 16 + row;
    const float bv = bias[d];
#pragma unroll
    for (int e = 0; e < 8; ++e) {
      const int s = s_base + e + half * 8;
      const float v = acc[t][e] + bv;
      if (mat == 2) {
        Vt[(((size_t)b * NH + h) * DH + d) * SEQ + s] = (__bf16)v;   // transposed
      } else {
        __bf16* dst = (mat == 0) ? Qb : Kb;
        dst[(((size_t)b * NH + h) * SEQ + s) * DH + d] = (__bf16)v;
      }
    }
  }
}

// ---------------------------------------------------------------- flash attention
// grid (S/128, B*H), block 256 = 8 waves, each wave owns a 16-row q-tile.
// K (32x64) and V (64x32) tiles staged cooperatively into LDS, double-buffered
// via async global->LDS DMA; all 8 waves share them (8x L2 traffic reduction).
__global__ __launch_bounds__(256) void attn(
    const __bf16* __restrict__ Qb, const __bf16* __restrict__ Kb,
    const __bf16* __restrict__ Vt, __bf16* __restrict__ wvb) {
  constexpr int KSTR = DH + 8;                   // bank-stagger padding
  constexpr int VSTR = 32 + 8;
  __shared__ alignas(16) __bf16 ksh[2][32 * KSTR];
  __shared__ alignas(16) __bf16 vsh[2][DH * VSTR];
  __shared__ alignas(16) __bf16 pb[8][16 * 32];  // per-wave P transpose staging

  const int tid = threadIdx.x, wave = tid >> 5, lane = tid & 31;
  const int half = lane >> 4, row = lane & 15;
  const int selA = half * 8, selB = half * 8;

  const int Q0 = blockIdx.x * 128;
  const int q_base = Q0 + wave * 16;
  const int bh = blockIdx.y, b = bh / NH, h = bh % NH;

  const __bf16* qbase = Qb + ((size_t)b * NH + h) * SEQ * DH;
  const __bf16* kbase = Kb + ((size_t)b * NH + h) * SEQ * DH;
  const __bf16* vbase = Vt + ((size_t)b * NH + h) * DH * SEQ;

  const __bf16* qrow = qbase + (size_t)(q_base + row) * DH;
  const v16bf aq0 = cat16(*(const v8bf*)(qrow +  0 + selA), *(const v8bf*)(qrow + 16 + selA));
  const v16bf aq1 = cat16(*(const v8bf*)(qrow + 32 + selA), *(const v8bf*)(qrow + 48 + selA));

  // cooperative K/V tile staging for step j into buffer bi
  const int kr = tid >> 3, kc = (tid & 7) * 8;   // K: 32 rows x 64 d
  const int vd = tid >> 2, vc = (tid & 3) * 8;   // V: 64 rows x 32 s
  auto stage = [&](int j, int bi) {
    const int kk0 = j * 32;
    cp16_g2l(kbase + (size_t)(kk0 + kr) * DH + kc, &ksh[bi][kr * KSTR + kc]);
    cp16_g2l(vbase + (size_t)vd * SEQ + kk0 + vc, &vsh[bi][vd * VSTR + vc]);
  };

  float m[8], l[8];
#pragma unroll
  for (int e = 0; e < 8; ++e) { m[e] = -__builtin_inff(); l[e] = 0.f; }
  v8f acc[4] = {};
  __bf16* myp = pb[wave];

  const int nsteps = Q0 / 32 + 4;                // covers kk < Q0+128 (block causal max)
  stage(0, 0);
  wait_async();
  __syncthreads();

  for (int j = 0; j < nsteps; ++j) {
    const int kk0 = j * 32;
    const int cur = j & 1;
    if (j + 1 < nsteps) stage(j + 1, cur ^ 1);   // prefetch overlaps compute

    const __bf16* kt = ksh[cur];
    const __bf16* vt = vsh[cur];

    v8f sc[2];
#pragma unroll
    for (int u = 0; u < 2; ++u) {
      const __bf16* krow = kt + (u * 16 + row) * KSTR;
      v16bf b0 = cat16(*(const v8bf*)(krow + selB),      *(const v8bf*)(krow + selB + 16));
      v16bf b1 = cat16(*(const v8bf*)(krow + 32 + selB), *(const v8bf*)(krow + 32 + selB + 16));
      v8f z = {};
      z = wmma_bf16(aq0, b0, z);
      z = wmma_bf16(aq1, b1, z);
      sc[u] = z;
    }
    float cf[8];
#pragma unroll
    for (int e = 0; e < 8; ++e) {
      const int q = q_base + e + half * 8;
      float v0 = sc[0][e] * 0.125f;              // 1/sqrt(64)
      float v1 = sc[1][e] * 0.125f;
      if (kk0 + row      > q) v0 = -100000.0f;   // IGNORE
      if (kk0 + 16 + row > q) v1 = -100000.0f;
      float rm = fmaxf(v0, v1);
      rm = fmaxf(rm, __shfl_xor(rm, 1, 32));
      rm = fmaxf(rm, __shfl_xor(rm, 2, 32));
      rm = fmaxf(rm, __shfl_xor(rm, 4, 32));
      rm = fmaxf(rm, __shfl_xor(rm, 8, 32));
      const float mn = fmaxf(m[e], rm);
      cf[e] = __expf(m[e] - mn);
      const float p0 = __expf(v0 - mn);
      const float p1 = __expf(v1 - mn);
      float rs = p0 + p1;
      rs += __shfl_xor(rs, 1, 32);
      rs += __shfl_xor(rs, 2, 32);
      rs += __shfl_xor(rs, 4, 32);
      rs += __shfl_xor(rs, 8, 32);
      l[e] = l[e] * cf[e] + rs;
      m[e] = mn;
      myp[(e + half * 8) * 32 + row]      = (__bf16)p0;   // C-layout -> LDS
      myp[(e + half * 8) * 32 + 16 + row] = (__bf16)p1;
    }
#pragma unroll
    for (int t = 0; t < 4; ++t)
#pragma unroll
      for (int e = 0; e < 8; ++e) acc[t][e] = acc[t][e] * cf[e];

    __builtin_amdgcn_wave_barrier();
    asm volatile("s_wait_dscnt 0" ::: "memory");   // same-wave LDS RAW only
    const v16bf ap = cat16(*(const v8bf*)(myp + row * 32 + selA),
                           *(const v8bf*)(myp + row * 32 + 16 + selA));
    __builtin_amdgcn_wave_barrier();
#pragma unroll
    for (int t = 0; t < 4; ++t) {
      const __bf16* vrow = vt + (t * 16 + row) * VSTR + selB;
      v16bf bv = cat16(*(const v8bf*)vrow, *(const v8bf*)(vrow + 16));
      acc[t] = wmma_bf16(ap, bv, acc[t]);
    }

    wait_async();      // prefetched tiles for j+1 have landed
    __syncthreads();   // safe to overwrite buffer cur in next prefetch
  }

#pragma unroll
  for (int t = 0; t < 4; ++t) {
    const int d = t * 16 + row;
#pragma unroll
    for (int e = 0; e < 8; ++e) {
      const int s = q_base + e + half * 8;
      wvb[((size_t)b * SEQ + s) * DM + h * DH + d] = (__bf16)(acc[t][e] / l[e]);
    }
  }
}

// ---------------------------------------------------------------- output projection
// grid (B*S/16, 2), block 256 (8 waves); wave -> 64 output columns.
// Ping-pong double buffering like qkv_proj.
__global__ __launch_bounds__(256) void out_proj(
    const __bf16* __restrict__ wvb, const __bf16* __restrict__ WoT,
    const float* __restrict__ bO, float* __restrict__ out) {
  const int tid = threadIdx.x, wave = tid >> 5, lane = tid & 31;
  const int half = lane >> 4, row = lane & 15;
  const int selA = half * 8, selB = half * 8;
  const int bs = blockIdx.x;
  const int b = bs / (SEQ / 16);
  const int s_base = (bs % (SEQ / 16)) * 16;
  const int mo_base = blockIdx.y * 512 + wave * 64;

  const __bf16* arow = wvb + ((size_t)b * SEQ + s_base + row) * DM;

  auto loadA = [&](int ks) -> v16bf {
    const int kb = ks * 32;
    return cat16(*(const v8bf*)(arow + kb + selA),
                 *(const v8bf*)(arow + kb + 16 + selA));
  };
  auto loadB = [&](int ks, int t) -> v16bf {
    const __bf16* wp = WoT + (size_t)(mo_base + t * 16 + row) * DM + ks * 32 + selB;
    return cat16(*(const v8bf*)wp, *(const v8bf*)(wp + 16));
  };

  constexpr int NK = DM / 32;                    // 32 k-steps (even)
  v8f acc[4] = {};
  v16bf a0 = loadA(0), a1;
  v16bf b0[4], b1[4];
#pragma unroll
  for (int t = 0; t < 4; ++t) b0[t] = loadB(0, t);

  for (int ks = 0; ks < NK - 2; ks += 2) {
    a1 = loadA(ks + 1);
#pragma unroll
    for (int t = 0; t < 4; ++t) b1[t] = loadB(ks + 1, t);
#pragma unroll
    for (int t = 0; t < 4; ++t) acc[t] = wmma_bf16(a0, b0[t], acc[t]);
    a0 = loadA(ks + 2);
#pragma unroll
    for (int t = 0; t < 4; ++t) b0[t] = loadB(ks + 2, t);
#pragma unroll
    for (int t = 0; t < 4; ++t) acc[t] = wmma_bf16(a1, b1[t], acc[t]);
  }
  a1 = loadA(NK - 1);
#pragma unroll
  for (int t = 0; t < 4; ++t) b1[t] = loadB(NK - 1, t);
#pragma unroll
  for (int t = 0; t < 4; ++t) acc[t] = wmma_bf16(a0, b0[t], acc[t]);
#pragma unroll
  for (int t = 0; t < 4; ++t) acc[t] = wmma_bf16(a1, b1[t], acc[t]);

#pragma unroll
  for (int t = 0; t < 4; ++t) {
    const int mo = mo_base + t * 16 + row;
    const float bias = bO[mo];
#pragma unroll
    for (int e = 0; e < 8; ++e) {
      const int s = s_base + e + half * 8;
      out[((size_t)b * SEQ + s) * DM + mo] = acc[t][e] + bias;
    }
  }
}

// ---------------------------------------------------------------- launch
extern "C" void kernel_launch(void* const* d_in, const int* in_sizes, int n_in,
                              void* d_out, int out_size, void* d_ws, size_t ws_size,
                              hipStream_t stream) {
  (void)in_sizes; (void)n_in; (void)out_size; (void)ws_size;
  const float* x  = (const float*)d_in[0];
  const float* Wq = (const float*)d_in[1];
  const float* Wk = (const float*)d_in[2];
  const float* Wv = (const float*)d_in[3];
  const float* Wo = (const float*)d_in[4];
  const float* bQ = (const float*)d_in[5];
  const float* bK = (const float*)d_in[6];
  const float* bV = (const float*)d_in[7];
  const float* bO = (const float*)d_in[8];
  float* out = (float*)d_out;

  char* ws = (char*)d_ws;
  const size_t qkvN = (size_t)BATCH * NH * SEQ * DH;   // 8M elements
  size_t o = 0;
  __bf16* Qb  = (__bf16*)(ws + o); o += qkvN * 2;
  __bf16* Kb  = (__bf16*)(ws + o); o += qkvN * 2;
  __bf16* Vt  = (__bf16*)(ws + o); o += qkvN * 2;
  __bf16* wvb = (__bf16*)(ws + o); o += (size_t)BATCH * SEQ * DM * 2;
  __bf16* WqT = (__bf16*)(ws + o); o += (size_t)NH * DH * DM * 2;
  __bf16* WkT = (__bf16*)(ws + o); o += (size_t)NH * DH * DM * 2;
  __bf16* WvT = (__bf16*)(ws + o); o += (size_t)NH * DH * DM * 2;
  __bf16* WoT = (__bf16*)(ws + o); o += (size_t)DM * DM * 2;

  const int prepN = 3 * NH * DH * DM + DM * DM;        // 4,194,304
  prep_weights<<<(prepN + 255) / 256, 256, 0, stream>>>(Wq, Wk, Wv, Wo, WqT, WkT, WvT, WoT);
  qkv_proj<<<dim3(BATCH * SEQ / 16, 6), 256, 0, stream>>>(x, WqT, WkT, WvT, bQ, bK, bV, Qb, Kb, Vt);
  attn<<<dim3(SEQ / 128, BATCH * NH), 256, 0, stream>>>(Qb, Kb, Vt, wvb);
  out_proj<<<dim3(BATCH * SEQ / 16, 2), 256, 0, stream>>>(wvb, WoT, bO, out);
}
